// StructuredStateRecurrence_13795434954809
// MI455X (gfx1250) — matrix operand order repaired and checked by reading, not compiled, not run on hardware
//
#include <hip/hip_runtime.h>
#include <hip/hip_bf16.h>

// ---------------------------------------------------------------------------
// StructuredStateRecurrence on MI455X (gfx1250, wave32, WMMA)
//
//   1) cvt x, Wv, Wg -> bf16 (x_bf16 = 67MB fits in 192MB L2)
//   2) GEMM1: v = x_bf16 @ Wv_bf16^T + bv          (v_wmma_f32_16x16x32_bf16)
//   3) scan over T per (b,m) channel -> wv_bf16, next_memory
//   4) GEMM2: out = x + [x_bf16 | wv_bf16] @ Wg_bf16^T + bg
//
// GEMM scheme: 256-thread block = 8 waves (2 row-groups x 4 col-groups).
// Block tile 128x256, wave tile 64x64 (16 WMMAs / 32-K-step).
// A tile (128x32 bf16) double-buffered in LDS (pitch 40 -> conflict-free),
// copied via global_load_async_to_lds_b128; the async copy of tile k+32
// overlaps the WMMA work on tile k (one s_wait_asynccnt + barrier per step).
// B read direct from L2-resident weights (contiguous 32B/lane).
// ---------------------------------------------------------------------------

typedef __attribute__((ext_vector_type(8)))  float  v8f;
typedef __attribute__((ext_vector_type(8)))  __bf16 v8bf;
typedef __attribute__((ext_vector_type(16))) __bf16 v16bf;
typedef int v4i __attribute__((vector_size(16)));   // matches builtin's V4i

#if __has_builtin(__builtin_amdgcn_global_load_async_to_lds_b128) && \
    __has_builtin(__builtin_amdgcn_s_wait_asynccnt)
#define HAVE_ASYNC_LDS 1
#else
#define HAVE_ASYNC_LDS 0
#endif

#define BB 4
#define TT 4096
#define DD 2048
#define MM 512
#define BT (BB*TT)      // 16384 rows
#define KG (DD+MM)      // 2560

#define APITCH   40     // LDS A-tile row pitch in bf16 (80B; bank stride 20)
#define BLK_ROWS 128
#define BLK_COLS 256
#define ATILE_ELEMS (BLK_ROWS * APITCH)

// copy 16 bf16 (32B) global -> LDS
__device__ __forceinline__ void copy16_to_lds(__bf16* dst, const __bf16* src) {
#if HAVE_ASYNC_LDS
  __builtin_amdgcn_global_load_async_to_lds_b128(
      (__attribute__((address_space(1))) v4i*)src,
      (__attribute__((address_space(3))) v4i*)dst, 0, 0);
  __builtin_amdgcn_global_load_async_to_lds_b128(
      (__attribute__((address_space(1))) v4i*)(src + 8),
      (__attribute__((address_space(3))) v4i*)(dst + 8), 0, 0);
#else
  uint4 t0 = *(const uint4*)(src);
  uint4 t1 = *(const uint4*)(src + 8);
  *(uint4*)(dst)     = t0;
  *(uint4*)(dst + 8) = t1;
#endif
}

// tile handoff: incoming async copy arrived + all waves done with other buffer
__device__ __forceinline__ void lds_tile_ready() {
#if HAVE_ASYNC_LDS
  __builtin_amdgcn_s_wait_asynccnt(0);
#endif
  __syncthreads();
}

// one 32-K MMA step: 4 B frags from global, 4 A frags from LDS, 16 WMMAs
__device__ __forceinline__ void mma_step(const __bf16* aldsbase,
                                         const __bf16* const brow[4],
                                         int kb, v8f acc[4][4]) {
  v16bf bfrag[4];
#pragma unroll
  for (int j = 0; j < 4; j++) {
    bfrag[j] = *(const v16bf*)(brow[j] + kb);
    __builtin_prefetch(brow[j] + kb + 32, 0, 3);  // speculative, safe past end
  }
  v16bf afrag[4];
#pragma unroll
  for (int i = 0; i < 4; i++) {
    const __bf16* p = aldsbase + i * 16 * APITCH;
    v8bf lo = *(const v8bf*)(p);
    v8bf hi = *(const v8bf*)(p + 8);
    afrag[i] = __builtin_shufflevector(lo, hi, 0,1,2,3,4,5,6,7,8,9,10,11,12,13,14,15);
  }
#pragma unroll
  for (int i = 0; i < 4; i++)
#pragma unroll
    for (int j = 0; j < 4; j++)
      acc[i][j] = __builtin_amdgcn_wmma_f32_16x16x32_bf16(
          false, afrag[i], false, bfrag[j], (short)0, acc[i][j], false, false);
}

// ---------------- f32 -> bf16, 8 elements per thread -----------------------
__global__ __launch_bounds__(256) void cvt_bf16_kernel(const float* __restrict__ in,
                                                       __bf16* __restrict__ out,
                                                       long long n8) {
  long long i = (long long)blockIdx.x * blockDim.x + threadIdx.x;
  if (i >= n8) return;
  float4 a = ((const float4*)in)[2 * i];
  float4 b = ((const float4*)in)[2 * i + 1];
  v8bf o;
  o[0] = (__bf16)a.x; o[1] = (__bf16)a.y; o[2] = (__bf16)a.z; o[3] = (__bf16)a.w;
  o[4] = (__bf16)b.x; o[5] = (__bf16)b.y; o[6] = (__bf16)b.z; o[7] = (__bf16)b.w;
  ((v8bf*)out)[i] = o;
}

// ---------------- GEMM1: v = x @ Wv^T + bv ---------------------------------
__global__ __launch_bounds__(256) void gemm_value_kernel(
    const __bf16* __restrict__ Xb,   // [BT, DD]
    const __bf16* __restrict__ Wvb,  // [MM, DD]
    const float*  __restrict__ bv,   // [MM]
    float* __restrict__ Vout)        // [BT, MM]
{
  __shared__ __bf16 Atile[2][ATILE_ELEMS];

  const int tid  = threadIdx.x;
  const int lane = tid & 31, wave = tid >> 5;
  const int mrow = lane & 15;       // A row / B col / C col within 16x16 tile
  const int hsel = lane >> 4;       // half-wave select
  const int waveRow = wave >> 2;    // 0..1
  const int blockRow = blockIdx.y * BLK_ROWS;
  const int colBase  = blockIdx.x * BLK_COLS + (wave & 3) * 64;

  v8f acc[4][4];
#pragma unroll
  for (int i = 0; i < 4; i++)
#pragma unroll
    for (int j = 0; j < 4; j++) acc[i][j] = (v8f){0,0,0,0,0,0,0,0};

  const __bf16* brow[4];
#pragma unroll
  for (int j = 0; j < 4; j++)
    brow[j] = Wvb + (size_t)(colBase + j * 16 + mrow) * DD;

  // cooperative A copy: thread -> (row, 32B segment)
  const int crow = tid >> 1;
  const int cseg = (tid & 1) * 16;
  const __bf16* asrc = Xb + (size_t)(blockRow + crow) * DD + cseg;
  __bf16* adst0 = &Atile[0][crow * APITCH + cseg];
  __bf16* adst1 = &Atile[1][crow * APITCH + cseg];

  // per-lane LDS read bases for A fragments
  const __bf16* albase0 = &Atile[0][(waveRow * 64 + mrow) * APITCH + hsel * 16];
  const __bf16* albase1 = &Atile[1][(waveRow * 64 + mrow) * APITCH + hsel * 16];

  copy16_to_lds(adst0, asrc);                    // prologue: tile 0 -> buf0

  for (int kt = 0; kt < DD; kt += 64) {
    lds_tile_ready();                            // buf0 = tile kt
    if (kt + 32 < DD) copy16_to_lds(adst1, asrc + kt + 32);
    mma_step(albase0, brow, kt + hsel * 16, acc);

    lds_tile_ready();                            // buf1 = tile kt+32
    if (kt + 64 < DD) copy16_to_lds(adst0, asrc + kt + 64);
    mma_step(albase1, brow, kt + 32 + hsel * 16, acc);
  }

#pragma unroll
  for (int j = 0; j < 4; j++) {
    const int col = colBase + j * 16 + mrow;
    const float bias = bv[col];
#pragma unroll
    for (int i = 0; i < 4; i++) {
      const int r0 = blockRow + waveRow * 64 + i * 16 + hsel * 8;
#pragma unroll
      for (int jj = 0; jj < 8; jj++)
        Vout[(size_t)(r0 + jj) * MM + col] = acc[i][j][jj] + bias;
    }
  }
}

// ---------------- scan over T per (b,m) channel ----------------------------
__global__ __launch_bounds__(256) void scan_kernel(
    const float* __restrict__ V,      // [BB,TT,MM]
    const float* __restrict__ s0,     // [BB,MM]
    const float* __restrict__ td,     // [MM]
    const float* __restrict__ tf,     // [MM]
    __bf16* __restrict__ WVb,         // [BB,TT,MM]
    float* __restrict__ nextMem)      // [BB,MM]
{
  const int idx = blockIdx.x * blockDim.x + threadIdx.x;
  if (idx >= BB * MM) return;
  const int b = idx / MM, m = idx % MM;
  const float decay = 1.0f / (1.0f + __expf(-td[m])) * 0.9f + 0.1f;
  const float first = 1.0f / (1.0f + __expf(-tf[m]));
  float s = s0[idx];
  const float* vp = V + (size_t)b * TT * MM + m;
  __bf16* wp = WVb + (size_t)b * TT * MM + m;
  for (int t = 0; t < TT; t++) {
    const float vt = vp[(size_t)t * MM];
    wp[(size_t)t * MM] = (__bf16)(vt + s * first);
    s = s * decay + vt;
  }
  nextMem[idx] = s;
}

// ---------------- GEMM2: out = x + [x|wv] @ Wg^T + bg ----------------------
__global__ __launch_bounds__(256) void gemm_gate_kernel(
    const __bf16* __restrict__ Xb,    // [BT, DD]
    const __bf16* __restrict__ WVb,   // [BT, MM]
    const __bf16* __restrict__ Wgb,   // [DD, KG]
    const float*  __restrict__ bg,    // [DD]
    const float*  __restrict__ Xf,    // [BT, DD] fp32 residual
    float* __restrict__ Out)          // [BT, DD]
{
  __shared__ __bf16 Atile[2][ATILE_ELEMS];

  const int tid  = threadIdx.x;
  const int lane = tid & 31, wave = tid >> 5;
  const int mrow = lane & 15;
  const int hsel = lane >> 4;
  const int waveRow = wave >> 2;
  const int blockRow = blockIdx.y * BLK_ROWS;
  const int colBase  = blockIdx.x * BLK_COLS + (wave & 3) * 64;

  v8f acc[4][4];
#pragma unroll
  for (int i = 0; i < 4; i++)
#pragma unroll
    for (int j = 0; j < 4; j++) acc[i][j] = (v8f){0,0,0,0,0,0,0,0};

  const __bf16* brow[4];
#pragma unroll
  for (int j = 0; j < 4; j++)
    brow[j] = Wgb + (size_t)(colBase + j * 16 + mrow) * KG;

  const int crow = tid >> 1;
  const int cseg = (tid & 1) * 16;
  const __bf16* asrcX = Xb  + (size_t)(blockRow + crow) * DD + cseg;
  const __bf16* asrcW = WVb + (size_t)(blockRow + crow) * MM + cseg;
  __bf16* adst0 = &Atile[0][crow * APITCH + cseg];
  __bf16* adst1 = &Atile[1][crow * APITCH + cseg];

  const __bf16* albase0 = &Atile[0][(waveRow * 64 + mrow) * APITCH + hsel * 16];
  const __bf16* albase1 = &Atile[1][(waveRow * 64 + mrow) * APITCH + hsel * 16];

  // concat-K source select (uniform: 2048 is a 32-multiple)
  #define A_SRC(k) ((k) < DD ? (asrcX + (k)) : (asrcW + ((k) - DD)))

  copy16_to_lds(adst0, A_SRC(0));                // prologue: tile 0 -> buf0

  for (int kt = 0; kt < KG; kt += 64) {
    lds_tile_ready();                            // buf0 = tile kt
    if (kt + 32 < KG) copy16_to_lds(adst1, A_SRC(kt + 32));
    mma_step(albase0, brow, kt + hsel * 16, acc);

    lds_tile_ready();                            // buf1 = tile kt+32
    if (kt + 64 < KG) copy16_to_lds(adst0, A_SRC(kt + 64));
    mma_step(albase1, brow, kt + 32 + hsel * 16, acc);
  }
  #undef A_SRC

#pragma unroll
  for (int j = 0; j < 4; j++) {
    const int col = colBase + j * 16 + mrow;
    const float bias = bg[col];
#pragma unroll
    for (int i = 0; i < 4; i++) {
      const int r0 = blockRow + waveRow * 64 + i * 16 + hsel * 8;
#pragma unroll
      for (int jj = 0; jj < 8; jj++) {
        const size_t o = (size_t)(r0 + jj) * DD + col;
        Out[o] = acc[i][j][jj] + Xf[o] + bias;
      }
    }
  }
}

// ---------------------------------------------------------------------------
extern "C" void kernel_launch(void* const* d_in, const int* in_sizes, int n_in,
                              void* d_out, int out_size, void* d_ws, size_t ws_size,
                              hipStream_t stream) {
  (void)in_sizes; (void)n_in; (void)out_size; (void)ws_size;
  const float* x    = (const float*)d_in[0];  // [B,T,D]
  const float* mem0 = (const float*)d_in[1];  // [B,M]
  const float* Wv   = (const float*)d_in[2];  // [M,D]
  const float* bv   = (const float*)d_in[3];  // [M]
  const float* Wg   = (const float*)d_in[4];  // [D,D+M]
  const float* bg   = (const float*)d_in[5];  // [D]
  const float* td   = (const float*)d_in[6];  // [M]
  const float* tf   = (const float*)d_in[7];  // [M]

  float* out     = (float*)d_out;                 // [B,T,D] flat
  float* nextMem = out + (size_t)BT * DD;         // then [B,M]

  char* ws = (char*)d_ws;
  __bf16* xb   = (__bf16*)ws;  ws += (size_t)BT * DD * sizeof(__bf16);
  __bf16* wvb  = (__bf16*)ws;  ws += (size_t)MM * DD * sizeof(__bf16);
  __bf16* wgb  = (__bf16*)ws;  ws += (size_t)DD * KG * sizeof(__bf16);
  float*  vbuf = (float*)ws;   ws += (size_t)BT * MM * sizeof(float);
  __bf16* wvab = (__bf16*)ws;  // [BT,MM] weighted values in bf16

  {
    long long n8x = (long long)BT * DD / 8;
    long long n8v = (long long)MM * DD / 8;
    long long n8g = (long long)DD * KG / 8;
    cvt_bf16_kernel<<<(int)((n8x + 255) / 256), 256, 0, stream>>>(x,  xb,  n8x);
    cvt_bf16_kernel<<<(int)((n8v + 255) / 256), 256, 0, stream>>>(Wv, wvb, n8v);
    cvt_bf16_kernel<<<(int)((n8g + 255) / 256), 256, 0, stream>>>(Wg, wgb, n8g);
  }

  gemm_value_kernel<<<dim3(MM / BLK_COLS, BT / BLK_ROWS), 256, 0, stream>>>(xb, wvb, bv, vbuf);

  scan_kernel<<<(BB * MM) / 256, 256, 0, stream>>>(vbuf, mem0, td, tf, wvab, nextMem);

  gemm_gate_kernel<<<dim3(DD / BLK_COLS, BT / BLK_ROWS), 256, 0, stream>>>(xb, wvab, wgb, bg, x, out);
}
